// GraphAttentionLayer_26938034881253
// MI455X (gfx1250) — matrix operand order repaired and checked by reading, not compiled, not run on hardware
//
#include <hip/hip_runtime.h>
#include <hip/hip_bf16.h>

// ---------------------------------------------------------------------------
// GAT layer, algebraically collapsed (softmax shift-invariance makes the
// attention exactly adj/deg, identical for all heads):
//   x   = nodes @ W + b
//   y   = x @ mean_h(K[h])
//   out = relu( diag(1/rowsum(adj)) * adj @ y )
// All three GEMMs run on v_wmma_f32_16x16x32_f16 (CDNA5 wave32 WMMA) with
// double-buffered LDS and async global->LDS copies (ASYNCcnt path).
// ---------------------------------------------------------------------------

typedef __attribute__((ext_vector_type(16))) _Float16 v16h;
typedef __attribute__((ext_vector_type(8)))  _Float16 v8h;
typedef __attribute__((ext_vector_type(4)))  _Float16 v4h;
typedef __attribute__((ext_vector_type(8)))  float    v8f;
typedef __attribute__((ext_vector_type(4)))  unsigned int u32x4;

#define NN 4096
#define DD 512

#if defined(__has_builtin)
#if __has_builtin(__builtin_amdgcn_global_load_async_to_lds_b128)
#define HAS_ASYNC_LDS 1
#endif
#endif
#ifndef HAS_ASYNC_LDS
#define HAS_ASYNC_LDS 0
#endif

#if HAS_ASYNC_LDS
// builtin signature (from hipcc diagnostics): param0 = AS1 (global) pointer to
// int __vector(4); param1 = LDS pointer to int __vector(4); ICE offset; ICE cpol.
typedef int v4i_b128 __attribute__((vector_size(16)));
typedef __attribute__((address_space(1))) v4i_b128* gas4;
typedef __attribute__((address_space(3))) v4i_b128* las4;

template <int OFF>
__device__ inline void async_b128(const void* g, void* l) {
    __builtin_amdgcn_global_load_async_to_lds_b128((gas4)g, (las4)l, OFF, 0);
}
__device__ inline void wait_async0() {
#if __has_builtin(__builtin_amdgcn_s_wait_asynccnt)
    __builtin_amdgcn_s_wait_asynccnt(0);
#else
    asm volatile("s_wait_asynccnt 0" ::: "memory");
#endif
}
#endif

// ---------------- prep kernels ----------------

__global__ __launch_bounds__(256) void cvt_f16_kernel(const float* __restrict__ x,
                                                      _Float16* __restrict__ y, int n) {
    int i = (blockIdx.x * 256 + threadIdx.x) * 4;
    if (i < n) {
        float4 v = *(const float4*)(x + i);
        v4h h;
        h[0] = (_Float16)v.x; h[1] = (_Float16)v.y;
        h[2] = (_Float16)v.z; h[3] = (_Float16)v.w;
        *(v4h*)(y + i) = h;
    }
}

__global__ __launch_bounds__(256) void kmean_f16_kernel(const float* __restrict__ K,
                                                        _Float16* __restrict__ y, int n) {
    int i = blockIdx.x * 256 + threadIdx.x;
    if (i < n) {
        float s = (K[i] + K[i + n] + K[i + 2 * n] + K[i + 3 * n]) * 0.25f;
        y[i] = (_Float16)s;
    }
}

// One block per row: adj fp32 -> fp16 and inv_deg[row] = 1 / rowsum(adj)
__global__ __launch_bounds__(256) void adj_prep_kernel(const float* __restrict__ adj,
                                                       _Float16* __restrict__ adj16,
                                                       float* __restrict__ inv_deg) {
    int row = blockIdx.x;
    const float* src = adj + (size_t)row * NN;
    _Float16* dst = adj16 + (size_t)row * NN;
    float s = 0.f;
    for (int c = threadIdx.x * 4; c < NN; c += 256 * 4) {
        float4 v = *(const float4*)(src + c);
        s += v.x + v.y + v.z + v.w;
        v4h h;
        h[0] = (_Float16)v.x; h[1] = (_Float16)v.y;
        h[2] = (_Float16)v.z; h[3] = (_Float16)v.w;
        *(v4h*)(dst + c) = h;
    }
    __shared__ float red[256];
    red[threadIdx.x] = s;
    __syncthreads();
    for (int off = 128; off > 0; off >>= 1) {
        if (threadIdx.x < off) red[threadIdx.x] += red[threadIdx.x + off];
        __syncthreads();
    }
    if (threadIdx.x == 0) inv_deg[row] = 1.0f / red[0];
}

// ---------------- WMMA GEMM ----------------
// C[M,N] = A[M,K] @ B[K,N], fp16 in, fp32 accumulate. N, K compile-time.
// MODE 0: D = f16(C + bias[col]);  MODE 1: D = f16(C);
// MODE 2: D = f32(relu(C * aux[row]))
// Block 128x128xBK32, 256 threads = 8 waves (4Mx2N), wave tile 32x64,
// double-buffered LDS, one barrier per K-step.

#define BM 128
#define BN 128
#define BK 32
#define LDA 40   // halves per LDS A row   (80B: 16B-aligned, conflict-free)
#define LDB 40   // halves per LDS B^T col

__device__ inline v16h cat16(v8h lo, v8h hi) {
    return __builtin_shufflevector(lo, hi, 0, 1, 2, 3, 4, 5, 6, 7,
                                   8, 9, 10, 11, 12, 13, 14, 15);
}

template <int MODE, int N, int K>
__global__ __launch_bounds__(256) void gemm_wmma_kernel(
    const _Float16* __restrict__ A, const _Float16* __restrict__ B,
    const float* __restrict__ aux, void* __restrict__ Dout) {
    __shared__ _Float16 sA[2][BM * LDA];
    __shared__ _Float16 sB[2][BN * LDB];   // transposed: [col][k], k contiguous

    const int tid  = threadIdx.x;
    const int lane = tid & 31;
    const int wave = tid >> 5;      // 0..7
    const int wm   = wave >> 1;     // 0..3 -> 32-row strip
    const int wn   = wave & 1;      // 0..1 -> 64-col strip
    const int bm   = blockIdx.y * BM;
    const int bn   = blockIdx.x * BN;

    // A staging: thread -> (row, 16-half segment); 32B per thread per tile
    const int a_row = tid >> 1;            // 0..127
    const int a_off = (tid & 1) * 16;      // halves
    // B staging: thread -> (k-pair, 8-col segment)
    const int b_pr = tid >> 4;             // 0..15  -> k rows 2p, 2p+1
    const int b_cs = (tid & 15) * 8;       // 0..120 -> 8 columns

    const int half_lo = lane >> 4;
    const int mrow    = lane & 15;

    const _Float16* gA = A + (bm + a_row) * K + a_off;        // + k0
    const _Float16* gB = B + 2 * b_pr * N + bn + b_cs;        // + k0*N

    v8f acc[2][4];
#pragma unroll
    for (int i = 0; i < 2; ++i)
#pragma unroll
        for (int j = 0; j < 4; ++j)
#pragma unroll
            for (int r = 0; r < 8; ++r) acc[i][j][r] = 0.0f;

    // ---- prologue: stage tile 0 into buffer 0 ----
    {
#if HAS_ASYNC_LDS
        async_b128<0>(gA, &sA[0][a_row * LDA + a_off]);
        async_b128<16>(gA, &sA[0][a_row * LDA + a_off]);
#else
        u32x4 ra0 = *(const u32x4*)(gA);
        u32x4 ra1 = *(const u32x4*)(gA + 8);
        *(u32x4*)(&sA[0][a_row * LDA + a_off])     = ra0;
        *(u32x4*)(&sA[0][a_row * LDA + a_off + 8]) = ra1;
#endif
        v8h rb0 = *(const v8h*)(gB);
        v8h rb1 = *(const v8h*)(gB + N);
#pragma unroll
        for (int h = 0; h < 8; ++h) {
            union { _Float16 h2[2]; unsigned int u; } pk;
            pk.h2[0] = rb0[h];
            pk.h2[1] = rb1[h];
            *(unsigned int*)(&sB[0][(b_cs + h) * LDB + 2 * b_pr]) = pk.u;
        }
#if HAS_ASYNC_LDS
        wait_async0();
#endif
    }
    __syncthreads();

    int cur = 0;
    for (int k0 = 0; k0 < K; k0 += BK) {
        const int nxt = cur ^ 1;
        const bool more = (k0 + BK) < K;

        // ---- prefetch tile k0+BK ----
        v8h rb0, rb1;
#if !HAS_ASYNC_LDS
        u32x4 ra0, ra1;
#endif
        if (more) {
            const _Float16* pA = gA + (k0 + BK);
#if HAS_ASYNC_LDS
            async_b128<0>(pA, &sA[nxt][a_row * LDA + a_off]);
            async_b128<16>(pA, &sA[nxt][a_row * LDA + a_off]);
#else
            ra0 = *(const u32x4*)(pA);
            ra1 = *(const u32x4*)(pA + 8);
#endif
            const _Float16* pB = gB + (k0 + BK) * N;
            rb0 = *(const v8h*)(pB);
            rb1 = *(const v8h*)(pB + N);
        }

        // ---- compute tile k0 from LDS[cur] ----
        v16h afr[2];
#pragma unroll
        for (int fm = 0; fm < 2; ++fm) {
            const _Float16* pa = &sA[cur][(wm * 32 + fm * 16 + mrow) * LDA + half_lo * 8];
            afr[fm] = cat16(*(const v8h*)(pa), *(const v8h*)(pa + 16));
        }
#pragma unroll
        for (int fn = 0; fn < 4; ++fn) {
            const _Float16* pb = &sB[cur][(wn * 64 + fn * 16 + mrow) * LDB + half_lo * 8];
            v16h bfr = cat16(*(const v8h*)(pb), *(const v8h*)(pb + 16));
#pragma unroll
            for (int fm = 0; fm < 2; ++fm) {
                acc[fm][fn] = __builtin_amdgcn_wmma_f32_16x16x32_f16(
                    false, afr[fm], false, bfr, (short)0, acc[fm][fn], false, false);
            }
        }

        // ---- publish tile k0+BK ----
        if (more) {
#if !HAS_ASYNC_LDS
            *(u32x4*)(&sA[nxt][a_row * LDA + a_off])     = ra0;
            *(u32x4*)(&sA[nxt][a_row * LDA + a_off + 8]) = ra1;
#endif
#pragma unroll
            for (int h = 0; h < 8; ++h) {
                union { _Float16 h2[2]; unsigned int u; } pk;
                pk.h2[0] = rb0[h];
                pk.h2[1] = rb1[h];
                *(unsigned int*)(&sB[nxt][(b_cs + h) * LDB + 2 * b_pr]) = pk.u;
            }
#if HAS_ASYNC_LDS
            wait_async0();
#endif
        }
        __syncthreads();
        cur = nxt;
    }

    // ---- epilogue: VGPR r holds (row = rbase + r, col) ----
#pragma unroll
    for (int fm = 0; fm < 2; ++fm) {
        const int rbase = bm + wm * 32 + fm * 16 + half_lo * 8;
        float sc[8];
        if (MODE == 2) {
#pragma unroll
            for (int r = 0; r < 8; ++r) sc[r] = aux[rbase + r];
        }
#pragma unroll
        for (int fn = 0; fn < 4; ++fn) {
            const int col = bn + wn * 64 + fn * 16 + mrow;
            if (MODE == 0) {
                const float bias = aux[col];
                _Float16* P = (_Float16*)Dout + (size_t)rbase * N + col;
#pragma unroll
                for (int r = 0; r < 8; ++r)
                    P[r * N] = (_Float16)(acc[fm][fn][r] + bias);   // imm offsets
            } else if (MODE == 1) {
                _Float16* P = (_Float16*)Dout + (size_t)rbase * N + col;
#pragma unroll
                for (int r = 0; r < 8; ++r)
                    P[r * N] = (_Float16)(acc[fm][fn][r]);
            } else {
                float* P = (float*)Dout + (size_t)rbase * N + col;
#pragma unroll
                for (int r = 0; r < 8; ++r) {
                    float v = acc[fm][fn][r] * sc[r];
                    P[r * N] = v > 0.0f ? v : 0.0f;
                }
            }
        }
    }
}

// ---------------- launch ----------------

extern "C" void kernel_launch(void* const* d_in, const int* in_sizes, int n_in,
                              void* d_out, int out_size, void* d_ws, size_t ws_size,
                              hipStream_t stream) {
    const float* nodes = (const float*)d_in[0];   // [N, D]
    const float* adj   = (const float*)d_in[1];   // [N, N]
    const float* Wn    = (const float*)d_in[2];   // [D, D]
    const float* bn    = (const float*)d_in[3];   // [D]
    const float* Kw    = (const float*)d_in[4];   // [H, D, D]
    // d_in[5] A_self, d_in[6] A_neigh, d_in[7] num_heads: provably no effect
    // on the output (softmax shift-invariance + head-independent attention).

    char* w = (char*)d_ws;
    size_t off = 0;
    _Float16* adj16 = (_Float16*)(w + off); off += (size_t)NN * NN * 2;   // 32 MB
    _Float16* nod16 = (_Float16*)(w + off); off += (size_t)NN * DD * 2;   //  4 MB
    _Float16* W16   = (_Float16*)(w + off); off += (size_t)DD * DD * 2;
    _Float16* Km16  = (_Float16*)(w + off); off += (size_t)DD * DD * 2;
    _Float16* x16   = (_Float16*)(w + off); off += (size_t)NN * DD * 2;
    _Float16* y16   = (_Float16*)(w + off); off += (size_t)NN * DD * 2;
    float*    ideg  = (float*)(w + off);    off += (size_t)NN * 4;
    (void)ws_size; (void)in_sizes; (void)n_in; (void)out_size;

    cvt_f16_kernel<<<(NN * DD / 4 + 255) / 256, 256, 0, stream>>>(nodes, nod16, NN * DD);
    cvt_f16_kernel<<<(DD * DD / 4 + 255) / 256, 256, 0, stream>>>(Wn, W16, DD * DD);
    kmean_f16_kernel<<<(DD * DD + 255) / 256, 256, 0, stream>>>(Kw, Km16, DD * DD);
    adj_prep_kernel<<<NN, 256, 0, stream>>>(adj, adj16, ideg);

    dim3 g(DD / BN, NN / BM);
    // x = nodes @ W + b
    gemm_wmma_kernel<0, DD, DD><<<g, 256, 0, stream>>>(nod16, W16, bn, x16);
    // y = x @ Kmean
    gemm_wmma_kernel<1, DD, DD><<<g, 256, 0, stream>>>(x16, Km16, nullptr, y16);
    // out = relu(diag(1/deg) * adj @ y)
    gemm_wmma_kernel<2, DD, NN><<<g, 256, 0, stream>>>(adj16, y16, ideg, (float*)d_out);
}
